// FatHydra_47201690583117
// MI455X (gfx1250) — matrix operand order repaired and checked by reading, not compiled, not run on hardware
//
#include <hip/hip_runtime.h>
#include <hip/hip_bf16.h>

// ---------------------------------------------------------------------------
// FatHydra on MI455X (gfx1250): bf16 WMMA trunk GEMM + grouped routed GEMM.
// h kept in bf16 in workspace (L2-resident, 33 MB << 192 MB L2).
// This round: Tensor Data Mover (TDM) staging for the trunk GEMM panels,
// ASYNCcnt DMA for the expert gather.
// ---------------------------------------------------------------------------

#define B_ROWS   8192
#define D_IN     1024
#define D_HID    2048
#define N_OUT    128
#define N_TABLES 16
#define MAX_TILES 544   // >= 8192/16 + 15 worst-case grouped tiles

typedef __attribute__((ext_vector_type(16))) __bf16 v16bf;
typedef __attribute__((ext_vector_type(8)))  float  v8f;
typedef __attribute__((ext_vector_type(4)))  unsigned int v4u;
typedef __attribute__((ext_vector_type(8)))  unsigned int v8u;

union Frag {
    v16bf v;
    uint4 q[2];
};

// ---- gfx1250 async global->LDS DMA (ASYNCcnt-tracked), 16B per lane -------
// ISA 15.18.3: GLOBAL_LOAD_ASYNC_TO_LDS_B128, VDST = LDS byte address VGPR.
// A generic pointer to LDS carries the LDS offset in its low 32 bits.
__device__ __forceinline__ void async_copy_b128(void* lds_dst, const void* gsrc) {
    unsigned int lds_off = (unsigned int)(unsigned long long)lds_dst;
    asm volatile("global_load_async_to_lds_b128 %0, %1, off"
                 :: "v"(lds_off), "v"(gsrc)
                 : "memory");
}

__device__ __forceinline__ void wait_async0() {
#if __has_builtin(__builtin_amdgcn_s_wait_asynccnt)
    __builtin_amdgcn_s_wait_asynccnt(0);
#else
    asm volatile("s_wait_asynccnt 0" ::: "memory");
#endif
}

// ---- gfx1250 Tensor Data Mover: 2D tile load, TENSORcnt-tracked -----------
// D# group0 (ISA 8.3): w0 = count=1; w1 = lds_addr; w2 = gaddr[31:0];
//                      w3 = gaddr[56:32] | type(2)<<30.
// D# group1 (ISA 8.4): built by caller (loop-invariant, wave-uniform).
__device__ __forceinline__ void tdm_load_tile(unsigned int lds_off,
                                              const void* gaddr, v8u g1) {
    unsigned long long ga = (unsigned long long)gaddr;
    v4u g0;
    g0[0] = 1u;                                       // count=1, user D#
    g0[1] = lds_off;                                  // LDS byte address
    g0[2] = (unsigned int)ga;                         // global_addr[31:0]
    g0[3] = (unsigned int)(ga >> 32) | (2u << 30);    // addr[56:32] | type=2
    asm volatile("tensor_load_to_lds %0, %1"
                 :: "s"(g0), "s"(g1)
                 : "memory");
}

// group1 for a [rows x 1024] bf16 tensor, 128x32 tile, LDS row pitch 80B:
// data_size=2B (1<<16), pad_enable (1<<20), pad_interval=64B (3<<22),
// pad_amount=4 DWORDs (3<<25); tensor_dim0=1024, tile_dim0=32, tile_dim1=128,
// tensor_dim0_stride=1024 elements.
__device__ __forceinline__ v8u tdm_group1_1024(unsigned int rows) {
    v8u g1;
    g1[0] = 0x06D10000u;
    g1[1] = 1024u << 16;           // tensor_dim0[15:0] << 16
    g1[2] = rows << 16;            // tensor_dim0[31:16]=0 | tensor_dim1[15:0]
    g1[3] = 32u << 16;             // tensor_dim1[31:16]=0 | tile_dim0=32
    g1[4] = 128u;                  // tile_dim1=128 | tile_dim2=0
    g1[5] = 1024u;                 // tensor_dim0_stride[31:0]
    g1[6] = 0u;                    // stride[47:32] | dim1_stride[15:0]
    g1[7] = 0u;                    // dim1_stride[47:16]
    return g1;
}

// --------------------------- f32 -> bf16 convert ---------------------------
__global__ void cvt_bf16(const float* __restrict__ s, __bf16* __restrict__ d, int n) {
    for (int i = blockIdx.x * blockDim.x + threadIdx.x; i < n;
         i += gridDim.x * blockDim.x) {
        d[i] = (__bf16)s[i];
    }
}

// ------------------------------- trunk GEMM --------------------------------
// h[8192,2048] = leaky_relu(x @ W1^T + b1); W1 row-major == B^T[N][K] layout.
// Block: 256 threads (8 waves), 128x128 C tile, K-panel of 32. TDM stages the
// A/B panels into double-buffered LDS while all 8 waves run WMMA.
__global__ __launch_bounds__(256) void trunk_gemm(
    const __bf16* __restrict__ xb,   // [8192,1024]
    const __bf16* __restrict__ w1b,  // [2048,1024]
    const float*  __restrict__ b1,   // [2048]
    __bf16*       __restrict__ hbf,  // [8192,2048]
    float*        __restrict__ route)// [8192]
{
    __shared__ __bf16 sA[2][128 * 40];  // 128 rows x 32 K, padded to 40 halves
    __shared__ __bf16 sB[2][128 * 40];

    const int tid  = threadIdx.x;
    const int lane = tid & 31;
    const int wave = tid >> 5;
    const int m0 = blockIdx.y * 128;
    const int n0 = blockIdx.x * 128;
    const int waveM = (wave & 3) * 32;   // 4 waves along M (32 rows each)
    const int waveN = (wave >> 2) * 64;  // 2 waves along N (64 cols each)

    v8f acc[2][4] = {};

    const int mrow = lane & 15;
    const int koA  = (lane >> 4) * 8;    // A: K half-offset 0 / 8
    const int k0B  = (lane >> 4) * 8;    // B: K half-offset 0 / 8
    const int colL = lane & 15;

    const v8u g1A = tdm_group1_1024(B_ROWS);  // x  tensor: 8192 rows
    const v8u g1B = tdm_group1_1024(D_HID);   // W1 tensor: 2048 rows

    // TDM-stage one 32-wide K panel of A and B into LDS buffer `buf`.
    auto stage = [&](int buf, int kk) {
        if (wave == 0) {
            tdm_load_tile((unsigned int)(unsigned long long)(void*)&sA[buf][0],
                          xb + (size_t)m0 * D_IN + kk, g1A);
            tdm_load_tile((unsigned int)(unsigned long long)(void*)&sB[buf][0],
                          w1b + (size_t)n0 * D_IN + kk, g1B);
        }
    };
    auto publish = [&]() {
        if (wave == 0) __builtin_amdgcn_s_wait_tensorcnt(0);
        __syncthreads();
    };

    stage(0, 0);
    publish();

    for (int kt = 0; kt < D_IN / 32; ++kt) {
        const int cur = kt & 1;
        if (kt + 1 < D_IN / 32) stage(cur ^ 1, (kt + 1) * 32);  // overlap DMA

        Frag a[2], b[4];
        #pragma unroll
        for (int i = 0; i < 2; ++i) {
            const uint4* p = reinterpret_cast<const uint4*>(
                sA[cur] + (waveM + i * 16 + mrow) * 40 + koA);
            a[i].q[0] = p[0];      // K = koA .. koA+7
            a[i].q[1] = p[2];      // K = koA+16 .. koA+23
        }
        #pragma unroll
        for (int j = 0; j < 4; ++j) {
            const uint4* p = reinterpret_cast<const uint4*>(
                sB[cur] + (waveN + j * 16 + colL) * 40 + k0B);
            b[j].q[0] = p[0]; b[j].q[1] = p[2];   // K = k0B..+7, k0B+16..+23
        }
        #pragma unroll
        for (int i = 0; i < 2; ++i)
            #pragma unroll
            for (int j = 0; j < 4; ++j)
                acc[i][j] = __builtin_amdgcn_wmma_f32_16x16x32_bf16(
                    false, a[i].v, false, b[j].v, (short)0, acc[i][j],
                    false, false);

        publish();   // next panel landed; all waves done reading `cur`
    }

    // Epilogue: bias + leaky_relu, store bf16 h (+ f32 routing column 2047).
    const int rOff = (lane >> 4) * 8;   // C layout: lanes 16-31 hold M = r+8
    #pragma unroll
    for (int i = 0; i < 2; ++i) {
        #pragma unroll
        for (int j = 0; j < 4; ++j) {
            const int gn = n0 + waveN + j * 16 + colL;
            const float bias = b1[gn];
            #pragma unroll
            for (int r = 0; r < 8; ++r) {
                const int gm = m0 + waveM + i * 16 + rOff + r;
                float v = acc[i][j][r] + bias;
                float hv = v > 0.0f ? v : 0.01f * v;
                hbf[(size_t)gm * D_HID + gn] = (__bf16)hv;
                if (gn == D_HID - 1) route[gm] = hv;
            }
        }
    }
}

// ------------------------------ routing plan -------------------------------
__global__ void zero_meta(int* counts, int* cursor, int* numTiles) {
    const int i = threadIdx.x;
    if (i < N_TABLES) { counts[i] = 0; cursor[i] = 0; }
    if (i == 0) *numTiles = 0;
}

__global__ void route_hist(const float* __restrict__ route,
                           int* __restrict__ idx, int* __restrict__ counts) {
    const int b = blockIdx.x * blockDim.x + threadIdx.x;
    if (b >= B_ROWS) return;
    int v = (int)route[b];
    v = v < 1 ? 1 : (v > N_TABLES ? N_TABLES : v);
    v -= 1;
    idx[b] = v;
    atomicAdd(&counts[v], 1);
}

__global__ void plan_tiles(const int* __restrict__ counts, int* __restrict__ offsets,
                           int* __restrict__ tileTable, int* __restrict__ tileStart,
                           int* __restrict__ tileCnt, int* __restrict__ numTiles) {
    if (threadIdx.x != 0 || blockIdx.x != 0) return;
    int off = 0, T = 0;
    for (int t = 0; t < N_TABLES; ++t) {
        offsets[t] = off;
        const int c = counts[t];
        const int nt = (c + 15) >> 4;
        for (int i = 0; i < nt; ++i) {
            tileTable[T] = t;
            tileStart[T] = off + i * 16;
            const int rem = c - i * 16;
            tileCnt[T] = rem < 16 ? rem : 16;
            ++T;
        }
        off += c;
    }
    offsets[N_TABLES] = off;
    *numTiles = T;
}

__global__ void scatter_rows(const int* __restrict__ idx,
                             const int* __restrict__ offsets,
                             int* __restrict__ cursor, int* __restrict__ order) {
    const int b = blockIdx.x * blockDim.x + threadIdx.x;
    if (b >= B_ROWS) return;
    const int t = idx[b];
    const int pos = offsets[t] + atomicAdd(&cursor[t], 1);
    order[pos] = b;
}

// --------------------------- grouped expert GEMM ---------------------------
// One block per 16-row tile sharing a table t: C[16 x 128] in WMMA,
// A = gathered h rows (async DMA into LDS), B = We[t] streamed from L2 with
// prefetch hints, fused bias+sigmoid.
__global__ __launch_bounds__(128) void expert_gemm(
    const __bf16* __restrict__ hbf,       // [8192,2048]
    const __bf16* __restrict__ web,       // [16,128,2048]
    const float*  __restrict__ be,        // [16,128]
    const int*    __restrict__ order,
    const int*    __restrict__ tileTable,
    const int*    __restrict__ tileStart,
    const int*    __restrict__ tileCnt,
    const int*    __restrict__ numTiles,
    float*        __restrict__ out)       // [8192,128]
{
    __shared__ __bf16 sH[16 * 2056];  // 16 gathered rows, padded to 2056 halves
    __shared__ int rowIds[16];

    const int bid = blockIdx.x;
    if (bid >= *numTiles) return;

    const int t     = tileTable[bid];
    const int start = tileStart[bid];
    const int cnt   = tileCnt[bid];
    const int tid  = threadIdx.x;
    const int lane = tid & 31;
    const int wave = tid >> 5;   // 4 waves, 32 cols each

    if (tid < 16) rowIds[tid] = order[start + (tid < cnt ? tid : 0)];
    __syncthreads();

    // Gather the 16 h rows into LDS via async DMA: 4096 16B chunks / 128 thr.
    for (int c = tid; c < 4096; c += 128) {
        const int row = c >> 8;           // c / 256
        const int off = (c & 255) * 8;    // half index within row
        async_copy_b128(sH + row * 2056 + off,
                        hbf + (size_t)rowIds[row] * D_HID + off);
    }
    wait_async0();
    __syncthreads();

    v8f acc[2] = {};
    const int colL = lane & 15;
    const int koA  = (lane >> 4) * 8;
    const int k0B  = (lane >> 4) * 8;
    const __bf16* wbase = web + (size_t)t * N_OUT * D_HID;

    for (int kk = 0; kk < D_HID; kk += 32) {
        Frag a;
        const uint4* pa = reinterpret_cast<const uint4*>(
            sH + (lane & 15) * 2056 + kk + koA);
        a.q[0] = pa[0];
        a.q[1] = pa[2];
        #pragma unroll
        for (int j = 0; j < 2; ++j) {
            const int col = wave * 32 + j * 16 + colL;
            const __bf16* gb = wbase + (size_t)col * D_HID + kk + k0B;
            __builtin_prefetch(gb + 256, 0, 1);   // pull K+256 into cache
            const uint4* pb = reinterpret_cast<const uint4*>(gb);
            Frag bfr; bfr.q[0] = pb[0]; bfr.q[1] = pb[2];
            acc[j] = __builtin_amdgcn_wmma_f32_16x16x32_bf16(
                false, a.v, false, bfr.v, (short)0, acc[j], false, false);
        }
    }

    const int rOff = (lane >> 4) * 8;
    #pragma unroll
    for (int j = 0; j < 2; ++j) {
        const int col = wave * 32 + j * 16 + colL;
        const float bias = be[t * N_OUT + col];
        #pragma unroll
        for (int r = 0; r < 8; ++r) {
            const int lm = rOff + r;
            if (lm < cnt) {
                const float v = acc[j][r] + bias;
                out[(size_t)rowIds[lm] * N_OUT + col] = 1.0f / (1.0f + __expf(-v));
            }
        }
    }
}

// ------------------------------- launch glue -------------------------------
extern "C" void kernel_launch(void* const* d_in, const int* in_sizes, int n_in,
                              void* d_out, int out_size, void* d_ws, size_t ws_size,
                              hipStream_t stream) {
    const float* x  = (const float*)d_in[0];
    const float* W1 = (const float*)d_in[1];
    const float* b1 = (const float*)d_in[2];
    const float* We = (const float*)d_in[3];
    const float* be = (const float*)d_in[4];
    float* out = (float*)d_out;

    char* ws = (char*)d_ws;
    size_t off = 0;
    auto alloc = [&](size_t bytes) -> void* {
        off = (off + 255) & ~(size_t)255;
        void* p = ws + off;
        off += bytes;
        return p;
    };

    __bf16* xb   = (__bf16*)alloc((size_t)B_ROWS * D_IN * 2);
    __bf16* w1b  = (__bf16*)alloc((size_t)D_HID * D_IN * 2);
    __bf16* web  = (__bf16*)alloc((size_t)N_TABLES * N_OUT * D_HID * 2);
    __bf16* hbf  = (__bf16*)alloc((size_t)B_ROWS * D_HID * 2);
    float*  route = (float*)alloc((size_t)B_ROWS * 4);
    int* idx      = (int*)alloc((size_t)B_ROWS * 4);
    int* order    = (int*)alloc((size_t)B_ROWS * 4);
    int* counts   = (int*)alloc(N_TABLES * 4);
    int* offsets  = (int*)alloc((N_TABLES + 1) * 4);
    int* cursor   = (int*)alloc(N_TABLES * 4);
    int* tileTable = (int*)alloc(MAX_TILES * 4);
    int* tileStart = (int*)alloc(MAX_TILES * 4);
    int* tileCnt   = (int*)alloc(MAX_TILES * 4);
    int* numTiles  = (int*)alloc(4);
    if (off > ws_size) return;  // workspace too small; bail deterministically

    // f32 -> bf16 conversions (one HBM pass each)
    cvt_bf16<<<2048, 256, 0, stream>>>(x,  xb,  B_ROWS * D_IN);
    cvt_bf16<<<1024, 256, 0, stream>>>(W1, w1b, D_HID * D_IN);
    cvt_bf16<<<1024, 256, 0, stream>>>(We, web, N_TABLES * N_OUT * D_HID);

    zero_meta<<<1, 32, 0, stream>>>(counts, cursor, numTiles);

    // Trunk GEMM: grid = (N tiles, M tiles)
    trunk_gemm<<<dim3(D_HID / 128, B_ROWS / 128), 256, 0, stream>>>(
        xb, w1b, b1, hbf, route);

    route_hist<<<B_ROWS / 256, 256, 0, stream>>>(route, idx, counts);
    plan_tiles<<<1, 1, 0, stream>>>(counts, offsets, tileTable, tileStart,
                                    tileCnt, numTiles);
    scatter_rows<<<B_ROWS / 256, 256, 0, stream>>>(idx, offsets, cursor, order);

    expert_gemm<<<MAX_TILES, 128, 0, stream>>>(
        hbf, web, be, order, tileTable, tileStart, tileCnt, numTiles, out);
}
